// SparseAttention_88562225643886
// MI455X (gfx1250) — matrix operand order repaired and checked by reading, not compiled, not run on hardware
//
#include <hip/hip_runtime.h>

// ---------------------------------------------------------------------------
// Sliding-window block attention for MI455X (gfx1250).
//   B=2 T=8192 C=1024 H=16 D=64 BS=64 LW=128 W=192 NB=128
// Pipeline: cvt(f32->bf16) -> 3x bf16 WMMA GEMM (async-LDS double buffered)
//           -> block attention (WMMA + reg softmax) -> bf16 WMMA GEMM (f32 out)
// ---------------------------------------------------------------------------

typedef __bf16 v16bf __attribute__((ext_vector_type(16)));
typedef __bf16 v8bf  __attribute__((ext_vector_type(8)));
typedef float  v8f   __attribute__((ext_vector_type(8)));
typedef int    v4i   __attribute__((ext_vector_type(4)));

#define Bb  2
#define Tt  8192
#define Cc  1024
#define Hh  16
#define Dd  64
#define BSs 64
#define LWw 128
#define Ww  192
#define NBb 128

// ---- CDNA5 async LDS staging (ASYNCcnt path), with sync fallback -----------

#if defined(__has_builtin)
#if __has_builtin(__builtin_amdgcn_global_load_async_to_lds_b128)
#define HAVE_ASYNC_LDS 1
#endif
#if __has_builtin(__builtin_amdgcn_s_wait_asynccnt)
#define HAVE_WAIT_ASYNC 1
#endif
#endif

// builtin signature (from hipcc diagnostic): param0 = global (addrspace 1)
// pointer to a 128-bit int vector, param1 = LDS (addrspace 3), then imm
// offset + imm cpol.
typedef __attribute__((address_space(1))) v4i* gp_b128;
typedef __attribute__((address_space(3))) v4i* lp_b128;

// 16-byte global -> LDS copy (async on gfx1250: global_load_async_to_lds_b128)
__device__ __forceinline__ void cp16(__bf16* dst_lds, const __bf16* src_glb) {
#if defined(HAVE_ASYNC_LDS)
  __builtin_amdgcn_global_load_async_to_lds_b128((gp_b128)src_glb,
                                                 (lp_b128)dst_lds, 0, 0);
#else
  *(v8bf*)dst_lds = *(const v8bf*)src_glb;
#endif
}

template <int N>
__device__ __forceinline__ void wait_async() {
#if defined(HAVE_ASYNC_LDS)
#if defined(HAVE_WAIT_ASYNC)
  __builtin_amdgcn_s_wait_asynccnt(N);
#else
  asm volatile("s_wait_asynccnt %0" ::"i"(N));
#endif
#endif
}

// ---- WMMA helpers ----------------------------------------------------------

__device__ __forceinline__ v8f wmma_bf16(v16bf a, v16bf b, v8f c) {
  return __builtin_amdgcn_wmma_f32_16x16x32_bf16(false, a, false, b, (short)0, c,
                                                 false, false);
}

// A fragment 16x32 bf16: lane(l,h): row=row0+l, K = h*8+{0..7} and h*8+16+{0..7}
__device__ __forceinline__ v16bf frag_a(const __bf16* base, int stride, int row0,
                                        int k0) {
  const int lane = threadIdx.x & 31;
  const int l = lane & 15, h = lane >> 4;
  const __bf16* p = base + (row0 + l) * stride + k0 + h * 8;
  v8bf lo = *(const v8bf*)(p);
  v8bf hi = *(const v8bf*)(p + 16);
  return __builtin_shufflevector(lo, hi, 0, 1, 2, 3, 4, 5, 6, 7, 8, 9, 10, 11, 12,
                                 13, 14, 15);
}

// B fragment 32x16 bf16 (operand stored [n][k], k contiguous):
// lane(l,h): col=col0+l, K = h*8+{0..7} and h*8+16+{0..7}
__device__ __forceinline__ v16bf frag_b(const __bf16* base, int stride, int col0,
                                        int k0) {
  const int lane = threadIdx.x & 31;
  const int l = lane & 15, h = lane >> 4;
  const __bf16* p = base + (col0 + l) * stride + k0 + h * 8;
  v8bf lo = *(const v8bf*)(p);
  v8bf hi = *(const v8bf*)(p + 16);
  return __builtin_shufflevector(lo, hi, 0, 1, 2, 3, 4, 5, 6, 7, 8, 9, 10, 11, 12,
                                 13, 14, 15);
}

// ---------------------------------------------------------------------------
// Kernel 0: f32 -> bf16 conversion (bandwidth pass; enables raw async staging)
// ---------------------------------------------------------------------------
__global__ __launch_bounds__(256) void cvt_kernel(const float* __restrict__ in,
                                                  __bf16* __restrict__ out,
                                                  long n) {
  long i = ((long)blockIdx.x * 256 + threadIdx.x) * 8;
  if (i + 8 <= n) {
    float4 a = *(const float4*)(in + i);
    float4 b = *(const float4*)(in + i + 4);
    v8bf o = {(__bf16)a.x, (__bf16)a.y, (__bf16)a.z, (__bf16)a.w,
              (__bf16)b.x, (__bf16)b.y, (__bf16)b.z, (__bf16)b.w};
    *(v8bf*)(out + i) = o;
  }
}

// ---------------------------------------------------------------------------
// GEMM: out = X(bf16, MxK) @ W^T(bf16, W is NxK row-major) + bias
//   WG 256 thr = 8 waves (4M x 2N), tile 128x128, K-step 32, wave tile 32x64.
//   Double-buffered async LDS staging.
//   MODE 0: dst bf16 head-major (B,H,T,D), value = (acc+bias)*scale
//   MODE 1: dst f32 row-major (M,N),      value = acc+bias
// ---------------------------------------------------------------------------
template <int MODE>
__global__ __launch_bounds__(256) void gemm_bf16(const __bf16* __restrict__ X,
                                                 const __bf16* __restrict__ W,
                                                 const float* __restrict__ bias,
                                                 void* __restrict__ dstv,
                                                 float scale) {
  __shared__ __align__(16) __bf16 sA[2][128 * 32];  // 2 x 8 KB
  __shared__ __align__(16) __bf16 sB[2][128 * 32];  // 2 x 8 KB

  const int tid  = threadIdx.x;
  const int lane = tid & 31, wave = tid >> 5;
  const int wm = wave & 3, wn = wave >> 2;  // 4 x 2 wave grid
  const int l = lane & 15, hf = lane >> 4;
  const long m0 = (long)blockIdx.x * 128;
  const int  n0 = blockIdx.y * 128;

  v8f acc[2][4] = {};

  // stage one 128x32 A tile + 128x32 B tile: 512+512 16B chunks, 4 per thread
  auto stage = [&](int buf, int k0) {
    for (int e = tid; e < 512; e += 256) {
      int r = e >> 2, c8 = (e & 3) * 8;
      cp16(&sA[buf][r * 32 + c8], &X[(size_t)(m0 + r) * Cc + k0 + c8]);
      cp16(&sB[buf][r * 32 + c8], &W[(size_t)(n0 + r) * Cc + k0 + c8]);
    }
  };

  stage(0, 0);
  const int KIT = Cc / 32;
  for (int it = 0; it < KIT; ++it) {
    const int buf = it & 1;
    if (it + 1 < KIT) {
      __syncthreads();                  // nobody still reading buf^1
      stage(buf ^ 1, (it + 1) * 32);    // prefetch next (4 async issues/thread)
      wait_async<4>();                  // current buf's 4 issues complete
    } else {
      wait_async<0>();
    }
    __syncthreads();                    // all waves' tiles visible

    v16bf af[2], bfr[4];
#pragma unroll
    for (int mi = 0; mi < 2; ++mi) af[mi] = frag_a(&sA[buf][0], 32, wm * 32 + mi * 16, 0);
#pragma unroll
    for (int ni = 0; ni < 4; ++ni) bfr[ni] = frag_b(&sB[buf][0], 32, wn * 64 + ni * 16, 0);
#pragma unroll
    for (int mi = 0; mi < 2; ++mi)
#pragma unroll
      for (int ni = 0; ni < 4; ++ni)
        acc[mi][ni] = wmma_bf16(af[mi], bfr[ni], acc[mi][ni]);
  }

  // epilogue
#pragma unroll
  for (int mi = 0; mi < 2; ++mi) {
#pragma unroll
    for (int ni = 0; ni < 4; ++ni) {
#pragma unroll
      for (int r = 0; r < 8; ++r) {
        long m = m0 + wm * 32 + mi * 16 + 8 * hf + r;  // row = b*T + t
        int  n = n0 + wn * 64 + ni * 16 + l;           // col = h*D + d
        float v = acc[mi][ni][r] + bias[n];
        if (MODE == 0) {
          v *= scale;
          int bb = (int)(m >> 13), t = (int)(m & (Tt - 1));
          int hd = n >> 6, dd = n & (Dd - 1);
          ((__bf16*)dstv)[(((size_t)bb * Hh + hd) * Tt + t) * Dd + dd] = (__bf16)v;
        } else {
          ((float*)dstv)[(size_t)m * Cc + n] = v;
        }
      }
    }
  }
}

// ---------------------------------------------------------------------------
// Attention: one WG (128 thr = 4 waves) per (b, h, block).
//   scores 64x192 = Q @ K^T (WMMA), mask, register softmax, out = P @ V (WMMA)
// ---------------------------------------------------------------------------
__global__ __launch_bounds__(128) void attn_kernel(
    const __bf16* __restrict__ q, const __bf16* __restrict__ k,
    const __bf16* __restrict__ v, __bf16* __restrict__ o) {
  __shared__ __align__(16) __bf16 sQ[BSs * Dd];     // 8 KB   [row][d]
  __shared__ __align__(16) __bf16 sKV[Ww * Dd];     // 24 KB  K:[w][d] -> V^T:[d][w]
  __shared__ __align__(16) __bf16 sP[4 * 16 * Ww];  // 24 KB

  const int nblk = blockIdx.x, hd = blockIdx.y, bb = blockIdx.z;
  const int bh   = bb * Hh + hd;
  const int tid  = threadIdx.x;
  const int lane = tid & 31, wave = tid >> 5;  // wave = 16-row M strip
  const int l = lane & 15, hf = lane >> 4;
  const int w0 = nblk * BSs - LWw / 2;

  const __bf16* qbase = q + ((size_t)bh * Tt + nblk * BSs) * Dd;
  const __bf16* kbase = k + (size_t)bh * Tt * Dd;
  const __bf16* vbase = v + (size_t)bh * Tt * Dd;

  // ---- async stage Q (512 chunks) and clamped K window (768 chunks) ----
  for (int e = tid; e < 512; e += 128) cp16(&sQ[e * 8], &qbase[e * 8]);
  for (int e = tid; e < 768; e += 128) {
    int w = e >> 2, c8 = (e & 3) * 8;
    int t = w0 + w;
    t = t < 0 ? 0 : (t > Tt - 1 ? Tt - 1 : t);
    cp16(&sKV[w * Dd + c8], &kbase[(size_t)t * Dd + c8]);
  }
  wait_async<0>();
  __syncthreads();

  // ---- scores: per wave a 16x192 strip (12 tiles x K64 -> 24 WMMA) ----
  v16bf a0 = frag_a(sQ, Dd, wave * 16, 0);
  v16bf a1 = frag_a(sQ, Dd, wave * 16, 32);
  v8f sc[12];
#pragma unroll
  for (int j = 0; j < 12; ++j) {
    v8f c = {};
    c = wmma_bf16(a0, frag_b(sKV, Dd, j * 16, 0), c);
    c = wmma_bf16(a1, frag_b(sKV, Dd, j * 16, 32), c);
    sc[j] = c;
  }
  __syncthreads();  // all waves done reading K -> reuse buffer for V^T

  // ---- stage V^T: sKV[d][w] = V[t(w)][d]  (16B global reads, b16 LDS scatter)
  for (int e = tid; e < 1536; e += 128) {
    int w = e >> 3, d8 = (e & 7) * 8;
    int t = w0 + w;
    t = t < 0 ? 0 : (t > Tt - 1 ? Tt - 1 : t);
    v8bf vv = *(const v8bf*)&vbase[(size_t)t * Dd + d8];
#pragma unroll
    for (int j = 0; j < 8; ++j) sKV[(d8 + j) * Ww + w] = vv[j];
  }

  // ---- mask + softmax in registers (col = 16j+l, row = 8*hf+r) ----
#pragma unroll
  for (int j = 0; j < 12; ++j) {
    int g = w0 + j * 16 + l;
    if (g < 0 || g >= Tt) {
#pragma unroll
      for (int r = 0; r < 8; ++r) sc[j][r] = -3.0e38f;
    }
  }
#pragma unroll
  for (int r = 0; r < 8; ++r) {
    float mx = -3.0e38f;
#pragma unroll
    for (int j = 0; j < 12; ++j) mx = fmaxf(mx, sc[j][r]);
#pragma unroll
    for (int off = 8; off; off >>= 1) mx = fmaxf(mx, __shfl_xor(mx, off, 32));
    float sum = 0.f;
#pragma unroll
    for (int j = 0; j < 12; ++j) {
      float p = __expf(sc[j][r] - mx);
      sc[j][r] = p;
      sum += p;
    }
#pragma unroll
    for (int off = 8; off; off >>= 1) sum += __shfl_xor(sum, off, 32);
    float inv = 1.0f / sum;
    int row = 8 * hf + r;
#pragma unroll
    for (int j = 0; j < 12; ++j)
      sP[(wave * 16 + row) * Ww + j * 16 + l] = (__bf16)(sc[j][r] * inv);
  }
  __syncthreads();  // V^T staged + all P strips written

  // ---- out = P(16x192) @ V(192x64): 4 N-tiles x 6 K-steps -> 24 WMMA ----
  const __bf16* myP = sP + wave * 16 * Ww;
  v16bf pa[6];
#pragma unroll
  for (int kk = 0; kk < 6; ++kk) pa[kk] = frag_a(myP, Ww, 0, kk * 32);
#pragma unroll
  for (int nj = 0; nj < 4; ++nj) {
    v8f oc = {};
#pragma unroll
    for (int kk = 0; kk < 6; ++kk)
      oc = wmma_bf16(pa[kk], frag_b(sKV, Ww, nj * 16, kk * 32), oc);
#pragma unroll
    for (int r = 0; r < 8; ++r) {
      int t  = nblk * BSs + wave * 16 + 8 * hf + r;
      int cc = hd * Dd + nj * 16 + l;
      o[((size_t)bb * Tt + t) * Cc + cc] = (__bf16)oc[r];
    }
  }
}

// ---------------------------------------------------------------------------
// Launch
// ---------------------------------------------------------------------------
extern "C" void kernel_launch(void* const* d_in, const int* in_sizes, int n_in,
                              void* d_out, int out_size, void* d_ws, size_t ws_size,
                              hipStream_t stream) {
  (void)in_sizes; (void)n_in; (void)out_size; (void)ws_size;

  const float* query = (const float*)d_in[0];
  const float* key   = (const float*)d_in[1];
  const float* value = (const float*)d_in[2];
  const float* Wq = (const float*)d_in[3];
  const float* bq = (const float*)d_in[4];
  const float* Wk = (const float*)d_in[5];
  const float* bk = (const float*)d_in[6];
  const float* Wv = (const float*)d_in[7];
  const float* bv = (const float*)d_in[8];
  const float* Wo = (const float*)d_in[9];
  const float* bo = (const float*)d_in[10];
  float* out = (float*)d_out;

  const size_t NE  = (size_t)Bb * Tt * Cc;  // 16,777,216 activation elems
  const size_t NW  = (size_t)Cc * Cc;       // 1,048,576 weight elems
  const size_t NE2 = NE * 2, NW2 = NW * 2;  // bytes (bf16)
  char* ws = (char*)d_ws;
  __bf16* xq = (__bf16*)(ws + 0 * NE2);
  __bf16* xk = (__bf16*)(ws + 1 * NE2);
  __bf16* xv = (__bf16*)(ws + 2 * NE2);
  __bf16* qh = (__bf16*)(ws + 3 * NE2);
  __bf16* kh = (__bf16*)(ws + 4 * NE2);
  __bf16* vh = (__bf16*)(ws + 5 * NE2);
  __bf16* at = (__bf16*)(ws + 6 * NE2);
  __bf16* wq = (__bf16*)(ws + 7 * NE2 + 0 * NW2);
  __bf16* wk = (__bf16*)(ws + 7 * NE2 + 1 * NW2);
  __bf16* wv = (__bf16*)(ws + 7 * NE2 + 2 * NW2);
  __bf16* wo = (__bf16*)(ws + 7 * NE2 + 3 * NW2);

  // 1) f32 -> bf16 conversion passes
  const int cvtA = (int)(NE / 2048), cvtW = (int)(NW / 2048);
  cvt_kernel<<<cvtA, 256, 0, stream>>>(query, xq, (long)NE);
  cvt_kernel<<<cvtA, 256, 0, stream>>>(key,   xk, (long)NE);
  cvt_kernel<<<cvtA, 256, 0, stream>>>(value, xv, (long)NE);
  cvt_kernel<<<cvtW, 256, 0, stream>>>(Wq, wq, (long)NW);
  cvt_kernel<<<cvtW, 256, 0, stream>>>(Wk, wk, (long)NW);
  cvt_kernel<<<cvtW, 256, 0, stream>>>(Wv, wv, (long)NW);
  cvt_kernel<<<cvtW, 256, 0, stream>>>(Wo, wo, (long)NW);

  // 2) QKV projections (head-major bf16 out; q pre-scaled by 1/sqrt(D))
  dim3 gp(Bb * Tt / 128, Cc / 128);  // 128 x 8
  gemm_bf16<0><<<gp, 256, 0, stream>>>(xq, wq, bq, qh, 0.125f);
  gemm_bf16<0><<<gp, 256, 0, stream>>>(xk, wk, bk, kh, 1.0f);
  gemm_bf16<0><<<gp, 256, 0, stream>>>(xv, wv, bv, vh, 1.0f);

  // 3) block attention
  attn_kernel<<<dim3(NBb, Hh, Bb), 128, 0, stream>>>(qh, kh, vh, at);

  // 4) output projection -> f32 (B,T,C)
  gemm_bf16<1><<<gp, 256, 0, stream>>>(at, wo, bo, out, 1.0f);
}
